// SS2DProcessor_6451040878784
// MI455X (gfx1250) — compile-verified, matrix operand order checked by reading, not run on hardware
//
#include <hip/hip_runtime.h>
#include <hip/hip_bf16.h>
#include <math.h>

// ---------------------------------------------------------------------------
// SS2D block for MI455X (gfx1250, wave32, WMMA).
// GEMMs run on v_wmma_f32_16x16x32_bf16 (fp32 accumulate); the selective scan
// is parallelized over (b,k) blocks x 1024 threads with LDS-staged step rows,
// using the CDNA5 async global->LDS path (ASYNCcnt) when the toolchain
// exposes it.
// ---------------------------------------------------------------------------

#define B_      4
#define HH      16
#define WW      16
#define DIM     128
#define D_INNER 128
#define D_STATE 256
#define DT_RANK 256
#define KDIR    4
#define LSEQ    256          // H*W
#define NTOK    (B_*LSEQ)    // 1024 tokens
#define EPS     1e-5f

typedef __attribute__((ext_vector_type(16))) __bf16 v16bf;
typedef __attribute__((ext_vector_type(8)))  float  v8f;

// CDNA5 async global->LDS (ASYNCcnt-tracked) support, compile-safe guarded.
#if defined(__HIP_DEVICE_COMPILE__) &&                                   \
    __has_builtin(__builtin_amdgcn_global_load_async_to_lds_b32) &&      \
    __has_builtin(__builtin_amdgcn_s_wait_asynccnt)
#define USE_ASYNC_LDS 1
#else
#define USE_ASYNC_LDS 0
#endif

#if USE_ASYNC_LDS
// Builtin signature (from clang diagnostics): param0 = int addrspace(1)*
// (printed '__device__ int *'), param1 = int addrspace(3)*, then imm offset
// and imm cpol. Build the pointers with int->pointer casts: flat LDS
// addresses carry the LDS byte offset in the low 32 bits (aperture rule),
// and global flat addresses are usable as AS1.
typedef __attribute__((address_space(3))) int lds_int_t;
typedef __attribute__((address_space(1))) int glb_int_t;
__device__ __forceinline__ lds_int_t* to_lds(void* p) {
  return (lds_int_t*)(unsigned int)(unsigned long long)p;
}
__device__ __forceinline__ glb_int_t* to_glb(const void* p) {
  return (glb_int_t*)(unsigned long long)p;
}
#endif

// ---------------------------------------------------------------------------
// WMMA fragment loader (16-bit operand, 16x32 tile, NT layout).
// lane = row (M or N) % 16 ; lane[4] selects K-half per the CDNA5 ISA layout:
//   VGPR 0..3 : K = 2v   + kh*8   (pairs)
//   VGPR 4..7 : K = 16+2v+ kh*8   (pairs)
// p points at (row0, k0) of a row-major, K-contiguous matrix with leading dim ld.
// ---------------------------------------------------------------------------
__device__ __forceinline__ v16bf load_frag_nt(const __bf16* __restrict__ p,
                                              int ld, int lane) {
  const int row = lane & 15;
  const int kh  = (lane >> 4) & 1;
  const __bf16* q = p + row * ld + kh * 8;
  v16bf f;
#pragma unroll
  for (int j = 0; j < 4; ++j) {
    f[2*j]     = q[2*j];
    f[2*j + 1] = q[2*j + 1];
    f[2*j + 8] = q[16 + 2*j];
    f[2*j + 9] = q[16 + 2*j + 1];
  }
  return f;
}

// fast sigmoid: v_exp_f32 + v_rcp_f32 (avoid the IEEE div expansion)
__device__ __forceinline__ float sigmoidf_(float x) {
  return __builtin_amdgcn_rcpf(1.0f + __expf(-x));
}
// fast softplus: v_exp_f32 + v_log_f32, overflow-guarded
__device__ __forceinline__ float softplusf_(float x) {
  return (x > 20.0f) ? x : __logf(1.0f + __expf(x));
}

// ---------------------------------------------------------------------------
// K0: f32 -> bf16 downcast (weights)
// ---------------------------------------------------------------------------
__global__ void cvt_bf16_kernel(const float* __restrict__ in,
                                __bf16* __restrict__ out, int n) {
  int i = blockIdx.x * blockDim.x + threadIdx.x;
  if (i < n) out[i] = (__bf16)in[i];
}

// ---------------------------------------------------------------------------
// K1: per-token LayerNorm over DIM=128, emit bf16 activations.
// grid = 1024 tokens, block = 128 threads (4 waves).
// ---------------------------------------------------------------------------
__global__ void ln_tokens_kernel(const float* __restrict__ x,
                                 __bf16* __restrict__ xln) {
  __shared__ float s1[4], s2[4];
  const int t = blockIdx.x;
  const int c = threadIdx.x;
  float v = x[t * DIM + c];
  float s = v, ss = v * v;
#pragma unroll
  for (int o = 16; o > 0; o >>= 1) {
    s  += __shfl_xor(s,  o, 32);
    ss += __shfl_xor(ss, o, 32);
  }
  const int wid = threadIdx.x >> 5;
  if ((threadIdx.x & 31) == 0) { s1[wid] = s; s2[wid] = ss; }
  __syncthreads();
  const float sum  = s1[0] + s1[1] + s1[2] + s1[3];
  const float sums = s2[0] + s2[1] + s2[2] + s2[3];
  const float mu   = sum * (1.0f / DIM);
  const float var  = sums * (1.0f / DIM) - mu * mu;
  const float r    = rsqrtf(var + EPS);
  xln[t * DIM + c] = (__bf16)((v - mu) * r);
}

// ---------------------------------------------------------------------------
// K2: in_proj WMMA GEMM. xz[t,e] = sum_d xln[t,d] * Win[e,d].
// M=1024 (tokens), N=256 (e), K=128. 1024 tiles of 16x16, 8 waves/block.
// Epilogue: e<128 -> x_in scattered to NCHW; e>=128 -> SiLU(z) stored.
// ---------------------------------------------------------------------------
__global__ void gemm_inproj_kernel(const __bf16* __restrict__ A,
                                   const __bf16* __restrict__ Wb,
                                   float* __restrict__ xin_nchw,
                                   float* __restrict__ zsilu) {
  const int lane = threadIdx.x & 31;
  const int wave = threadIdx.x >> 5;
  const int tile = blockIdx.x * 8 + wave;     // 64*16 = 1024 tiles
  const int tm = tile >> 4;                   // 0..63
  const int tn = tile & 15;                   // 0..15
  v8f acc = {};
#pragma unroll
  for (int kk = 0; kk < 128; kk += 32) {
    v16bf a = load_frag_nt(A  + (tm * 16) * 128 + kk, 128, lane);
    v16bf b = load_frag_nt(Wb + (tn * 16) * 128 + kk, 128, lane);
    acc = __builtin_amdgcn_wmma_f32_16x16x32_bf16(false, a, false, b,
                                                  (short)0, acc, false, false);
  }
  const int n  = lane & 15;
  const int mh = lane >> 4;
  const int e  = tn * 16 + n;
#pragma unroll
  for (int r = 0; r < 8; ++r) {
    const int t  = tm * 16 + r + mh * 8;
    const float v = acc[r];
    const int bb = t >> 8, l = t & 255;
    if (e < D_INNER) {
      xin_nchw[(bb * D_INNER + e) * LSEQ + l] = v;          // NCHW for conv
    } else {
      zsilu[t * D_INNER + (e - D_INNER)] = v * sigmoidf_(v); // SiLU(z)
    }
  }
}

// ---------------------------------------------------------------------------
// K3: depthwise 3x3 conv (pad 1) + bias + SiLU, then scatter the value into
// the 4 directional sequences, stored time-major (b,k,l,c):
//   k=0: l=h*16+w   k=1: 255-l   k=2: lv=w*16+h   k=3: 255-lv
// Emits both f32 (scan) and bf16 (WMMA B operand) copies.
// grid = B*128 channels, block = 256 (one thread per pixel).
// ---------------------------------------------------------------------------
__global__ void conv_scatter_kernel(const float* __restrict__ xin_nchw,
                                    const float* __restrict__ conv_w,
                                    const float* __restrict__ conv_b,
                                    float* __restrict__ xsf,
                                    __bf16* __restrict__ xsT) {
  const int bc = blockIdx.x;
  const int b  = bc / D_INNER;
  const int c  = bc % D_INNER;
  const int h  = threadIdx.x >> 4;
  const int w  = threadIdx.x & 15;
  const float* src = xin_nchw + (b * D_INNER + c) * LSEQ;
  const float* wk  = conv_w + c * 9;
  float acc = conv_b[c];
#pragma unroll
  for (int dy = -1; dy <= 1; ++dy) {
#pragma unroll
    for (int dx = -1; dx <= 1; ++dx) {
      const int yy = h + dy, xx = w + dx;
      if (yy >= 0 && yy < HH && xx >= 0 && xx < WW)
        acc += wk[(dy + 1) * 3 + (dx + 1)] * src[yy * 16 + xx];
    }
  }
  const float v = acc * sigmoidf_(acc);   // SiLU
  const int l0 = h * 16 + w;
  const int lv = w * 16 + h;
  const int lidx[KDIR] = { l0, 255 - l0, lv, 255 - lv };
#pragma unroll
  for (int k = 0; k < KDIR; ++k) {
    const size_t idx = ((size_t)(b * KDIR + k) * LSEQ + lidx[k]) * D_INNER + c;
    xsf[idx] = v;
    xsT[idx] = (__bf16)v;
  }
}

// ---------------------------------------------------------------------------
// K4: x_dbl WMMA GEMM per (b,k): [768x256] = Wp_k[768x128] * xs_bk[128x256].
// NT form: A = Wp_k (d rows, c contig), B = xsT (l rows, c contig).
// Epilogue scatters time-major: dt_rT (bf16, l x r), Bs/Cs (f32, l x n).
// grid = (96, 16), block = 256.
// ---------------------------------------------------------------------------
__global__ void gemm_xdbl_kernel(const __bf16* __restrict__ Wp,
                                 const __bf16* __restrict__ XsT,
                                 __bf16* __restrict__ dtrT,
                                 float* __restrict__ Bsl,
                                 float* __restrict__ Csl) {
  const int bk = blockIdx.y;
  const int k  = bk & 3;
  const __bf16* A  = Wp  + (size_t)k  * 768 * 128;
  const __bf16* Bm = XsT + (size_t)bk * LSEQ * 128;
  const int lane = threadIdx.x & 31;
  const int wave = threadIdx.x >> 5;
  const int tile = blockIdx.x * 8 + wave;     // 48*16 = 768 tiles
  const int tm = tile >> 4;                   // 0..47
  const int tn = tile & 15;                   // 0..15
  v8f acc = {};
#pragma unroll
  for (int kk = 0; kk < 128; kk += 32) {
    v16bf a = load_frag_nt(A  + (tm * 16) * 128 + kk, 128, lane);
    v16bf b = load_frag_nt(Bm + (tn * 16) * 128 + kk, 128, lane);
    acc = __builtin_amdgcn_wmma_f32_16x16x32_bf16(false, a, false, b,
                                                  (short)0, acc, false, false);
  }
  const int n  = lane & 15;
  const int mh = lane >> 4;
  const int l  = tn * 16 + n;
  const size_t base = (size_t)bk * LSEQ * 256;
#pragma unroll
  for (int r = 0; r < 8; ++r) {
    const int d   = tm * 16 + r + mh * 8;
    const float v = acc[r];
    if (d < DT_RANK)                 dtrT[base + (size_t)l * 256 + d]              = (__bf16)v;
    else if (d < DT_RANK + D_STATE)  Bsl [base + (size_t)l * 256 + (d - DT_RANK)]  = v;
    else                             Csl [base + (size_t)l * 256 + (d - DT_RANK - D_STATE)] = v;
  }
}

// ---------------------------------------------------------------------------
// K5: dt WMMA GEMM per (b,k): [128x256] = Wdt_k[128x256] * dt_r[256x256].
// NT form: A = Wdt (c rows, r contig), B = dt_rT (l rows, r contig).
// Epilogue: + bias, softplus, store time-major dts(b,k,l,c).
// grid = (16, 16), block = 256.
// ---------------------------------------------------------------------------
__global__ void gemm_dt_kernel(const __bf16* __restrict__ Wdt,
                               const __bf16* __restrict__ dtrT,
                               const float* __restrict__ dtb,
                               float* __restrict__ dts) {
  const int bk = blockIdx.y;
  const int k  = bk & 3;
  const __bf16* A  = Wdt  + (size_t)k  * D_INNER * DT_RANK;
  const __bf16* Bm = dtrT + (size_t)bk * LSEQ * DT_RANK;
  const int lane = threadIdx.x & 31;
  const int wave = threadIdx.x >> 5;
  const int tile = blockIdx.x * 8 + wave;     // 8*16 = 128 tiles
  const int tm = tile >> 4;                   // 0..7
  const int tn = tile & 15;                   // 0..15
  v8f acc = {};
#pragma unroll
  for (int kk = 0; kk < DT_RANK; kk += 32) {
    v16bf a = load_frag_nt(A  + (tm * 16) * DT_RANK + kk, DT_RANK, lane);
    v16bf b = load_frag_nt(Bm + (tn * 16) * DT_RANK + kk, DT_RANK, lane);
    acc = __builtin_amdgcn_wmma_f32_16x16x32_bf16(false, a, false, b,
                                                  (short)0, acc, false, false);
  }
  const int n  = lane & 15;
  const int mh = lane >> 4;
  const int l  = tn * 16 + n;
#pragma unroll
  for (int r = 0; r < 8; ++r) {
    const int c = tm * 16 + r + mh * 8;
    const float v = acc[r] + dtb[k * D_INNER + c];
    dts[(size_t)bk * LSEQ * D_INNER + (size_t)l * D_INNER + c] = softplusf_(v);
  }
}

// ---------------------------------------------------------------------------
// K6: selective scan. One block per (b,k); 1024 threads.
// thread t: channel c = t>>3, state slice q = t&7 -> n in [q*32, q*32+32).
// h[32], A[32] live in VGPRs. Per step: B_t/C_t/dt/u staged in LDS via the
// CDNA5 async global->LDS path (ASYNCcnt) when available.
// 8-lane shfl_xor reduction (lanes sharing c are consecutive).
// ys(b,k,l,c) = scan_y + Ds*u.
// ---------------------------------------------------------------------------
__global__ void __launch_bounds__(1024)
scan_kernel(const float* __restrict__ Bsl, const float* __restrict__ Csl,
            const float* __restrict__ dts, const float* __restrict__ xsf,
            const float* __restrict__ A_log, const float* __restrict__ Ds,
            float* __restrict__ ys) {
  __shared__ float sB[D_STATE], sC[D_STATE], sdt[D_INNER], su[D_INNER];
  const int bk = blockIdx.x;
  const int k  = bk & 3;
  const int t  = threadIdx.x;
  const int c  = t >> 3;
  const int q  = t & 7;
  const int n0 = q * 32;

  const float* pB  = Bsl + (size_t)bk * LSEQ * D_STATE;
  const float* pC  = Csl + (size_t)bk * LSEQ * D_STATE;
  const float* pdt = dts + (size_t)bk * LSEQ * D_INNER;
  const float* pu  = xsf + (size_t)bk * LSEQ * D_INNER;
  float*       py  = ys  + (size_t)bk * LSEQ * D_INNER;

  float a[32], h[32];
#pragma unroll
  for (int j = 0; j < 32; ++j) {
    a[j] = -__expf(A_log[((size_t)k * D_INNER + c) * D_STATE + n0 + j]);
    h[j] = 0.0f;
  }
  const float dsv = Ds[k * D_INNER + c];

  for (int l = 0; l < LSEQ; ++l) {
    __syncthreads();
#if USE_ASYNC_LDS
    // GLOBAL_LOAD_ASYNC_TO_LDS_B32: memory -> LDS, no VGPR round trip.
    if (t < 256)
      __builtin_amdgcn_global_load_async_to_lds_b32(
          to_glb(pB + (size_t)l * D_STATE + t), to_lds(&sB[t]), 0, 0);
    else if (t < 512)
      __builtin_amdgcn_global_load_async_to_lds_b32(
          to_glb(pC + (size_t)l * D_STATE + (t - 256)), to_lds(&sC[t - 256]), 0, 0);
    else if (t < 640)
      __builtin_amdgcn_global_load_async_to_lds_b32(
          to_glb(pdt + (size_t)l * D_INNER + (t - 512)), to_lds(&sdt[t - 512]), 0, 0);
    else if (t < 768)
      __builtin_amdgcn_global_load_async_to_lds_b32(
          to_glb(pu + (size_t)l * D_INNER + (t - 640)), to_lds(&su[t - 640]), 0, 0);
    __builtin_amdgcn_s_wait_asynccnt(0);
#else
    if (t < 256)      sB [t]       = pB [(size_t)l * D_STATE + t];
    else if (t < 512) sC [t - 256] = pC [(size_t)l * D_STATE + (t - 256)];
    else if (t < 640) sdt[t - 512] = pdt[(size_t)l * D_INNER + (t - 512)];
    else if (t < 768) su [t - 640] = pu [(size_t)l * D_INNER + (t - 640)];
#endif
    if (t >= 768 && l + 1 < LSEQ) {
      // pull next step's rows toward the WGP (global_prefetch_b8)
      if (t == 768) __builtin_prefetch(pB  + (size_t)(l + 1) * D_STATE, 0, 1);
      if (t == 769) __builtin_prefetch(pC  + (size_t)(l + 1) * D_STATE, 0, 1);
      if (t == 770) __builtin_prefetch(pdt + (size_t)(l + 1) * D_INNER, 0, 1);
      if (t == 771) __builtin_prefetch(pu  + (size_t)(l + 1) * D_INNER, 0, 1);
    }
    __syncthreads();
    const float dt  = sdt[c];
    const float u   = su[c];
    const float dtu = dt * u;
    float acc = 0.0f;
#pragma unroll
    for (int j = 0; j < 32; ++j) {
      const float dA = __expf(dt * a[j]);
      h[j] = fmaf(dA, h[j], dtu * sB[n0 + j]);
      acc  = fmaf(h[j], sC[n0 + j], acc);
    }
    acc += __shfl_xor(acc, 1, 32);
    acc += __shfl_xor(acc, 2, 32);
    acc += __shfl_xor(acc, 4, 32);
    if (q == 0) py[(size_t)l * D_INNER + c] = acc + dsv * u;
  }
}

// ---------------------------------------------------------------------------
// K7: merge 4 directions, LayerNorm over c, gamma/beta, gate with SiLU(z).
// grid = 1024 tokens (b*256+l), block = 128.
// ---------------------------------------------------------------------------
__global__ void merge_ln_gate_kernel(const float* __restrict__ ys,
                                     const float* __restrict__ gamma,
                                     const float* __restrict__ beta,
                                     const float* __restrict__ zsilu,
                                     float* __restrict__ yfin) {
  __shared__ float s1[4], s2[4];
  const int tok = blockIdx.x;
  const int b   = tok >> 8;
  const int l   = tok & 255;
  const int c   = threadIdx.x;
  const int h   = l >> 4, w = l & 15;
  const int lv  = w * 16 + h;

  const size_t bs = (size_t)b * KDIR * LSEQ * D_INNER;
  float m = ys[bs + ((size_t)0 * LSEQ + l)         * D_INNER + c]
          + ys[bs + ((size_t)1 * LSEQ + (255 - l))  * D_INNER + c]
          + ys[bs + ((size_t)2 * LSEQ + lv)         * D_INNER + c]
          + ys[bs + ((size_t)3 * LSEQ + (255 - lv)) * D_INNER + c];

  float s = m, ss = m * m;
#pragma unroll
  for (int o = 16; o > 0; o >>= 1) {
    s  += __shfl_xor(s,  o, 32);
    ss += __shfl_xor(ss, o, 32);
  }
  const int wid = threadIdx.x >> 5;
  if ((threadIdx.x & 31) == 0) { s1[wid] = s; s2[wid] = ss; }
  __syncthreads();
  const float sum  = s1[0] + s1[1] + s1[2] + s1[3];
  const float sums = s2[0] + s2[1] + s2[2] + s2[3];
  const float mu   = sum * (1.0f / D_INNER);
  const float var  = sums * (1.0f / D_INNER) - mu * mu;
  const float r    = rsqrtf(var + EPS);
  float v = (m - mu) * r * gamma[c] + beta[c];
  v *= zsilu[(size_t)tok * D_INNER + c];
  yfin[(size_t)tok * D_INNER + c] = v;
}

// ---------------------------------------------------------------------------
// K8: deterministic spatial mean: out[b,c] = mean_l yfin[b,l,c].
// grid = B, block = 128.
// ---------------------------------------------------------------------------
__global__ void mean_kernel(const float* __restrict__ yfin,
                            float* __restrict__ out) {
  const int b = blockIdx.x;
  const int c = threadIdx.x;
  float s = 0.0f;
  for (int l = 0; l < LSEQ; ++l)
    s += yfin[((size_t)b * LSEQ + l) * D_INNER + c];
  out[b * D_INNER + c] = s * (1.0f / LSEQ);
}

// ---------------------------------------------------------------------------
// Host launcher
// ---------------------------------------------------------------------------
extern "C" void kernel_launch(void* const* d_in, const int* in_sizes, int n_in,
                              void* d_out, int out_size, void* d_ws, size_t ws_size,
                              hipStream_t stream) {
  const float* x        = (const float*)d_in[0];   // (4,16,16,128)
  const float* in_pw    = (const float*)d_in[1];   // (256,128)
  const float* conv_w   = (const float*)d_in[2];   // (128,1,3,3)
  const float* conv_b   = (const float*)d_in[3];   // (128)
  const float* x_proj_w = (const float*)d_in[4];   // (4,768,128)
  const float* dt_pw    = (const float*)d_in[5];   // (4,128,256)
  const float* dt_pb    = (const float*)d_in[6];   // (4,128)
  const float* A_log    = (const float*)d_in[7];   // (4,128,256)
  const float* Ds       = (const float*)d_in[8];   // (4,128)
  const float* gamma    = (const float*)d_in[9];   // (128)
  const float* beta     = (const float*)d_in[10];  // (128)
  float* out = (float*)d_out;                      // (4,128)

  // workspace carve-out (~20 MB), 256B aligned
  char* ws = (char*)d_ws;
  size_t off = 0;
  auto carve = [&](size_t bytes) -> char* {
    char* p = ws + off;
    off += (bytes + 255) & ~(size_t)255;
    return p;
  };
  __bf16* xln     = (__bf16*)carve((size_t)NTOK * DIM * 2);
  __bf16* win_b   = (__bf16*)carve((size_t)2 * D_INNER * DIM * 2);
  __bf16* wproj_b = (__bf16*)carve((size_t)KDIR * 768 * 128 * 2);
  __bf16* wdt_b   = (__bf16*)carve((size_t)KDIR * D_INNER * DT_RANK * 2);
  float*  xin     = (float* )carve((size_t)B_ * D_INNER * LSEQ * 4);
  float*  zsilu   = (float* )carve((size_t)NTOK * D_INNER * 4);
  float*  xsf     = (float* )carve((size_t)B_ * KDIR * LSEQ * D_INNER * 4);
  __bf16* xsT     = (__bf16*)carve((size_t)B_ * KDIR * LSEQ * D_INNER * 2);
  __bf16* dtrT    = (__bf16*)carve((size_t)B_ * KDIR * LSEQ * DT_RANK * 2);
  float*  Bsl     = (float* )carve((size_t)B_ * KDIR * LSEQ * D_STATE * 4);
  float*  Csl     = (float* )carve((size_t)B_ * KDIR * LSEQ * D_STATE * 4);
  float*  dts     = (float* )carve((size_t)B_ * KDIR * LSEQ * D_INNER * 4);
  float*  ys      = (float* )carve((size_t)B_ * KDIR * LSEQ * D_INNER * 4);
  float*  yfin    = (float* )carve((size_t)NTOK * D_INNER * 4);
  (void)ws_size; (void)n_in; (void)in_sizes; (void)out_size;

  // weight downcasts
  cvt_bf16_kernel<<<(2*D_INNER*DIM + 255) / 256, 256, 0, stream>>>(in_pw, win_b, 2*D_INNER*DIM);
  cvt_bf16_kernel<<<(KDIR*768*128 + 255) / 256, 256, 0, stream>>>(x_proj_w, wproj_b, KDIR*768*128);
  cvt_bf16_kernel<<<(KDIR*D_INNER*DT_RANK + 255) / 256, 256, 0, stream>>>(dt_pw, wdt_b, KDIR*D_INNER*DT_RANK);

  // LN + in_proj (WMMA) + conv/scatter
  ln_tokens_kernel<<<NTOK, 128, 0, stream>>>(x, xln);
  gemm_inproj_kernel<<<128, 256, 0, stream>>>(xln, win_b, xin, zsilu);
  conv_scatter_kernel<<<B_ * D_INNER, 256, 0, stream>>>(xin, conv_w, conv_b, xsf, xsT);

  // x_dbl (WMMA) and dt projection (WMMA)
  gemm_xdbl_kernel<<<dim3(96, B_ * KDIR), 256, 0, stream>>>(wproj_b, xsT, dtrT, Bsl, Csl);
  gemm_dt_kernel<<<dim3(16, B_ * KDIR), 256, 0, stream>>>(wdt_b, dtrT, dt_pb, dts);

  // selective scan, merge, gate, reduce
  scan_kernel<<<B_ * KDIR, 1024, 0, stream>>>(Bsl, Csl, dts, xsf, A_log, Ds, ys);
  merge_ln_gate_kernel<<<NTOK, 128, 0, stream>>>(ys, gamma, beta, zsilu, yfin);
  mean_kernel<<<B_, 128, 0, stream>>>(yfin, out);
}
